// PIGNN_12146167513810
// MI455X (gfx1250) — matrix operand order repaired and checked by reading, not compiled, not run on hardware
//
#include <hip/hip_runtime.h>

// ---------------------------------------------------------------------------
// CDNA5 (gfx1250) wave32 WMMA implementation of the PIGNN reference.
// v_wmma_f32_16x16x32_bf16 GEMMs with double-buffered LDS pipeline, fused
// gather/concat on A, fused bias+CELU / LayerNorm / residual / edge-update /
// atomic scatter epilogues.
// ---------------------------------------------------------------------------

typedef __attribute__((ext_vector_type(16))) __bf16 v16bf;
typedef __attribute__((ext_vector_type(8)))  __bf16 v8bf;
typedef __attribute__((ext_vector_type(8)))  float  v8f;
typedef __attribute__((ext_vector_type(4)))  float  v4f;

#define F_CELU 1
#define F_LN   2
#define F_RES  4
#define F_EDGE 8
#define F_SCAT 16

// ---------------------------------------------------------------------------
// Fused MLP-layer GEMM:  C[M,256] = epilogue( gatherA[M,K] @ W[K,256] + bias )
// A = concat of up to 3 segments [0,k1) [k1,k2) [k2,K), each with optional
// row-index indirection. Tile: 64 rows x 256 cols per 256-thread block;
// 8 waves, each owning a 32x64 sub-tile = 2x4 wmma accumulators.
// K-loop is software pipelined with ping-pong LDS buffers.
// ---------------------------------------------------------------------------
__global__ __launch_bounds__(256) void k_wmma_mlp(
    int M, int K, int k1, int k2,
    const float* __restrict__ s0, const int* __restrict__ i0, int ld0,
    const float* __restrict__ s1, const int* __restrict__ i1, int ld1,
    const float* __restrict__ s2, const int* __restrict__ i2, int ld2,
    const float* __restrict__ W,  const float* __restrict__ bias,
    int flags,
    const float* __restrict__ ln_g, const float* __restrict__ ln_b,
    float* __restrict__ out_f32,
    const float* residual,
    float* __restrict__ e_fwd, float* __restrict__ e_bwd,
    const int* __restrict__ dstI, const int* __restrict__ srcI,
    float* __restrict__ agg)
{
    __shared__ __align__(16) __bf16 Asb[2][64][40];   // ping-pong 64x32 bf16 A
    __shared__ __align__(16) __bf16 Bsb[2][256][40];  // ping-pong 256col x 32K (transposed)
    __shared__ float Cs[64][260];                      // f32 output tile (+pad)
    __shared__ float redS[64][4];
    __shared__ float redQ[64][4];

    const int tid   = threadIdx.x;
    const int lane  = tid & 31;
    const int wave  = tid >> 5;
    const int wm    = wave >> 2;      // row half of tile (0..1)
    const int wn    = wave & 3;       // col quarter (0..3)
    const int lhalf = lane >> 4;      // 0/1
    const int l16   = lane & 15;
    const int rowBase = blockIdx.x * 64;

    // ---- A-staging role: thread -> (row, 8-wide K chunk); hoist gather rows
    const int ar  = tid >> 2;            // 0..63
    const int akq = (tid & 3) * 8;       // 0,8,16,24
    const int grow_a = rowBase + ar;
    long r0 = 0, r1 = 0, r2 = 0;
    if (grow_a < M) {
        int j0 = i0 ? i0[grow_a] : grow_a;  r0 = (long)j0 * ld0;
        if (s1) { int j1 = i1 ? i1[grow_a] : grow_a; r1 = (long)j1 * ld1; }
        if (s2) { int j2 = i2 ? i2[grow_a] : grow_a; r2 = (long)j2 * ld2; }
    }
    // all segment boundaries multiple of 8 -> whole 8-chunk in one segment
    const bool fast8 = ((k1 | k2 | K) & 7) == 0;

    // ---- B-staging role: thread -> (K pair, 16 cols)
    const int bk = (tid >> 4) * 2;       // even K 0..30
    const int bc = (tid & 15) * 16;      // col block 0..240

    // ---- register-tile fetch (global only; no LDS)
    auto fetchA = [&](int kk, v8bf& aP) {
        const int gk0 = kk + akq;
        if (fast8) {
            const float* p = nullptr;
            if (grow_a < M) {
                if (gk0 < k1)      p = s0 + r0 + gk0;
                else if (gk0 < k2) p = s1 + r1 + (gk0 - k1);
                else if (gk0 < K)  p = s2 + r2 + (gk0 - k2);
            }
            v4f a = {}, b = {};
            if (p) { a = *(const v4f*)p; b = *(const v4f*)(p + 4); }
            #pragma unroll
            for (int t = 0; t < 4; ++t) {
                aP[t]     = (__bf16)a[t];
                aP[4 + t] = (__bf16)b[t];
            }
        } else {
            #pragma unroll
            for (int j = 0; j < 8; ++j) {
                int gk = gk0 + j;
                float v = 0.f;
                if (grow_a < M) {
                    if (gk < k1)      v = s0[r0 + gk];
                    else if (gk < k2) v = s1[r1 + (gk - k1)];
                    else if (gk < K)  v = s2[r2 + (gk - k2)];
                }
                aP[j] = (__bf16)v;
            }
        }
    };
    auto fetchB = [&](int kk, unsigned (&bp)[16]) {
        const bool in0 = (kk + bk)     < K;
        const bool in1 = (kk + bk + 1) < K;
        const float* p0 = W + (long)(kk + bk) * 256 + bc;
        const float* p1 = p0 + 256;
        #pragma unroll
        for (int j = 0; j < 16; j += 4) {
            v4f a = {}, b = {};
            if (in0) a = *(const v4f*)(p0 + j);
            if (in1) b = *(const v4f*)(p1 + j);
            #pragma unroll
            for (int t = 0; t < 4; ++t) {
                unsigned short lo = __builtin_bit_cast(unsigned short, (__bf16)a[t]);
                unsigned short hi = __builtin_bit_cast(unsigned short, (__bf16)b[t]);
                bp[j + t] = ((unsigned)hi << 16) | lo;
            }
        }
    };
    auto storeTile = [&](int buf, const v8bf& aP, const unsigned (&bp)[16]) {
        *(v8bf*)&Asb[buf][ar][akq] = aP;                 // ds_store_b128
        #pragma unroll
        for (int j = 0; j < 16; ++j)
            *(unsigned*)&Bsb[buf][bc + j][bk] = bp[j];   // packed b32 stores
    };

    v8f acc[2][4];
    {
        v8f zero = {};
        #pragma unroll
        for (int mi = 0; mi < 2; ++mi)
            #pragma unroll
            for (int ni = 0; ni < 4; ++ni) acc[mi][ni] = zero;
    }

    const int nsteps = (K + 31) >> 5;
    {
        v8bf aP; unsigned bp[16];
        fetchA(0, aP); fetchB(0, bp);
        storeTile(0, aP, bp);
    }
    __syncthreads();

    for (int s = 0; s < nsteps; ++s) {
        const int cur = s & 1;
        v8bf aP; unsigned bp[16];
        const bool more = (s + 1) < nsteps;
        if (more) { fetchA((s + 1) << 5, aP); fetchB((s + 1) << 5, bp); }

        // ---- WMMA from buffer `cur` (A 16x32 / B 32x16 bf16 ISA layouts)
        v16bf afrag[2];
        #pragma unroll
        for (int mi = 0; mi < 2; ++mi) {
            const __bf16* p = &Asb[cur][wm * 32 + mi * 16 + l16][lhalf * 8];
            v8bf lo = *(const v8bf*)p;          // K 0..7   (or 8..15)
            v8bf hi = *(const v8bf*)(p + 16);   // K 16..23 (or 24..31)
            afrag[mi] = __builtin_shufflevector(lo, hi,
                0,1,2,3,4,5,6,7,8,9,10,11,12,13,14,15);
        }
        #pragma unroll
        for (int ni = 0; ni < 4; ++ni) {
            const __bf16* p = &Bsb[cur][wn * 64 + ni * 16 + l16][lhalf * 8];
            v8bf lo = *(const v8bf*)p;
            v8bf hi = *(const v8bf*)(p + 16);
            v16bf bfrag = __builtin_shufflevector(lo, hi,
                0,1,2,3,4,5,6,7,8,9,10,11,12,13,14,15);
            #pragma unroll
            for (int mi = 0; mi < 2; ++mi)
                acc[mi][ni] = __builtin_amdgcn_wmma_f32_16x16x32_bf16(
                    false, afrag[mi], false, bfrag, (short)0, acc[mi][ni],
                    false, false);
        }

        if (more) storeTile(cur ^ 1, aP, bp);   // fill the other buffer
        __syncthreads();
    }

    // ---- epilogue: bias (+CELU) into LDS f32 tile (full 256-wide rows)
    #pragma unroll
    for (int mi = 0; mi < 2; ++mi)
        #pragma unroll
        for (int ni = 0; ni < 4; ++ni) {
            const int cl = wn * 64 + ni * 16 + l16;
            const float bv = bias[cl];
            #pragma unroll
            for (int v = 0; v < 8; ++v) {
                const int rl = wm * 32 + mi * 16 + lhalf * 8 + v;
                float x = acc[mi][ni][v] + bv;
                if (flags & F_CELU) x = (x > 0.f) ? x : expm1f(x);
                Cs[rl][cl] = x;
            }
        }
    __syncthreads();

    const int er = tid >> 2;   // row 0..63
    const int eq = tid & 3;    // column quarter
    float mu = 0.f, rstd = 1.f;
    if (flags & F_LN) {
        float s = 0.f, q = 0.f;
        #pragma unroll
        for (int j = 0; j < 64; j += 4) {
            v4f x = *(const v4f*)&Cs[er][eq * 64 + j];
            #pragma unroll
            for (int t = 0; t < 4; ++t) { s += x[t]; q += x[t] * x[t]; }
        }
        redS[er][eq] = s; redQ[er][eq] = q;
        __syncthreads();
        float ts = redS[er][0] + redS[er][1] + redS[er][2] + redS[er][3];
        float tq = redQ[er][0] + redQ[er][1] + redQ[er][2] + redQ[er][3];
        mu = ts * (1.f / 256.f);
        float var = tq * (1.f / 256.f) - mu * mu;
        rstd = rsqrtf(var + 1e-5f);
    }

    const int grow = rowBase + er;
    if (grow >= M) return;
    const long obase = (long)grow * 256;
    int dI = 0, sI = 0;
    if (flags & F_SCAT) { dI = dstI[grow]; sI = srcI[grow]; }
    #pragma unroll 4
    for (int j = 0; j < 64; j += 4) {
        const int c = eq * 64 + j;
        v4f x = *(const v4f*)&Cs[er][c];
        if (flags & F_LN) {
            v4f g4 = *(const v4f*)&ln_g[c];
            v4f b4 = *(const v4f*)&ln_b[c];
            #pragma unroll
            for (int t = 0; t < 4; ++t) x[t] = (x[t] - mu) * rstd * g4[t] + b4[t];
        }
        if (flags & F_RES) {
            v4f r4 = *(const v4f*)&residual[obase + c];
            x += r4;
        }
        if (out_f32) *(v4f*)&out_f32[obase + c] = x;
        if (flags & F_EDGE) {
            v4f f4 = *(const v4f*)&e_fwd[obase + c];
            v4f w4 = *(const v4f*)&e_bwd[obase + c];
            f4 += x; w4 -= x;
            *(v4f*)&e_fwd[obase + c] = f4;
            *(v4f*)&e_bwd[obase + c] = w4;
        }
        if (flags & F_SCAT) {
            #pragma unroll
            for (int t = 0; t < 4; ++t) {
                atomicAdd(&agg[(long)dI * 256 + c + t],  x[t]);  // global_atomic_add_f32
                atomicAdd(&agg[(long)sI * 256 + c + t], -x[t]);
            }
        }
    }
}

// incoming[idx[j]] += src[j]  over `rows` 256-wide rows (one wave per row)
__global__ __launch_bounds__(256) void k_scatter_rows(
    const float* __restrict__ src, const int* __restrict__ idx,
    float* __restrict__ outacc, int rows)
{
    const int j = blockIdx.x * 8 + (threadIdx.x >> 5);
    if (j >= rows) return;
    const int lane = threadIdx.x & 31;
    const int n = idx[j];
    const float* sp = src + (long)j * 256;
    float* op = outacc + (long)n * 256;
    #pragma unroll
    for (int u = 0; u < 8; ++u) {
        const int c = lane + 32 * u;
        atomicAdd(&op[c], sp[c]);
    }
}

// z[n] = LN_512( concat(h[n], inc[n]) ) * g + b      (one block per node)
__global__ __launch_bounds__(256) void k_final_ln(
    const float* __restrict__ h, const float* __restrict__ inc,
    const float* __restrict__ g, const float* __restrict__ b,
    float* __restrict__ z, int N)
{
    __shared__ float sm[256], sq[256];
    const int n = blockIdx.x;
    const int t = threadIdx.x;
    const float x0 = h  [(long)n * 256 + t];
    const float x1 = inc[(long)n * 256 + t];
    sm[t] = x0 + x1; sq[t] = x0 * x0 + x1 * x1;
    __syncthreads();
    for (int off = 128; off > 0; off >>= 1) {
        if (t < off) { sm[t] += sm[t + off]; sq[t] += sq[t + off]; }
        __syncthreads();
    }
    const float mu   = sm[0] * (1.f / 512.f);
    const float var  = sq[0] * (1.f / 512.f) - mu * mu;
    const float rstd = rsqrtf(var + 1e-5f);
    z[(long)n * 512 + t]       = (x0 - mu) * rstd * g[t]       + b[t];
    z[(long)n * 512 + 256 + t] = (x1 - mu) * rstd * g[256 + t] + b[256 + t];
}

// decoder last layer: out[n*3+comp] = (dot256(hid[n], w) + b1[0]) * (1-mask[n])
__global__ __launch_bounds__(256) void k_dec_dot(
    const float* __restrict__ hid, const float* __restrict__ w,
    const float* __restrict__ b1, const float* __restrict__ mask,
    float* __restrict__ out, int N, int comp)
{
    const int n = blockIdx.x * 8 + (threadIdx.x >> 5);
    if (n >= N) return;
    const int lane = threadIdx.x & 31;
    const float* hp = hid + (long)n * 256;
    float s = 0.f;
    #pragma unroll
    for (int u = 0; u < 8; ++u) {
        const int c = lane + 32 * u;
        s += hp[c] * w[c];
    }
    #pragma unroll
    for (int m = 16; m > 0; m >>= 1) s += __shfl_xor(s, m, 32);
    if (lane == 0)
        out[(long)n * 3 + comp] = (s + b1[0]) * (1.f - mask[n]);
}

// ---------------------------------------------------------------------------
extern "C" void kernel_launch(void* const* d_in, const int* in_sizes, int n_in,
                              void* d_out, int out_size, void* d_ws, size_t ws_size,
                              hipStream_t stream)
{
    const int N  = in_sizes[0] / 10;   // x: [N,10]
    const int E  = in_sizes[1] / 14;   // edge_attr: [2E,7]
    const int E2 = 2 * E;

    const float* x         = (const float*)d_in[0];
    const float* edge_attr = (const float*)d_in[1];
    const float* bc_disp   = (const float*)d_in[2];
    const float* bc_rot    = (const float*)d_in[3];

    struct MLP { const float *W0, *W1, *b0, *b1, *g, *b; };
    auto P = [&](int i){ return (const float*)d_in[i]; };
    MLP node_enc, edge_enc, mp_e[6], mp_n[6], dec[3];
    const float *fin_g, *fin_b;

    // Detect flatten convention: insertion-order dict flatten puts
    // node_enc.W0 (10*256=2560) at index 4; jax sorted-key tree_leaves puts
    // dec_th.W0 (512*256=131072) there.
    if (in_sizes[4] == 2560) {
        node_enc = { P(4), P(5), P(6), P(7), P(8), P(9) };
        edge_enc = { P(10), P(11), P(12), P(13), P(14), P(15) };
        for (int l = 0; l < 6; ++l) {
            int eb = 16 + 12 * l, nb = eb + 6;
            mp_e[l] = { P(eb), P(eb+1), P(eb+2), P(eb+3), P(eb+4), P(eb+5) };
            mp_n[l] = { P(nb), P(nb+1), P(nb+2), P(nb+3), P(nb+4), P(nb+5) };
        }
        fin_g = P(88); fin_b = P(89);
        dec[0] = { P(90), P(91), P(92), P(93), 0, 0 };   // ux
        dec[1] = { P(94), P(95), P(96), P(97), 0, 0 };   // uz
        dec[2] = { P(98), P(99), P(100), P(101), 0, 0 }; // th
    } else {
        dec[2] = { P(4),  P(5),  P(6),  P(7),  0, 0 };   // dec_th
        dec[0] = { P(8),  P(9),  P(10), P(11), 0, 0 };   // dec_ux
        dec[1] = { P(12), P(13), P(14), P(15), 0, 0 };   // dec_uz
        edge_enc = { P(16), P(17), P(18), P(19), P(21), P(20) }; // ln_b before ln_g
        fin_b = P(22); fin_g = P(23);
        for (int l = 0; l < 6; ++l) {
            int eb = 24 + 12 * l, nb = eb + 6;
            mp_e[l] = { P(eb), P(eb+1), P(eb+2), P(eb+3), P(eb+5), P(eb+4) };
            mp_n[l] = { P(nb), P(nb+1), P(nb+2), P(nb+3), P(nb+5), P(nb+4) };
        }
        node_enc = { P(96), P(97), P(98), P(99), P(101), P(100) };
    }
    const int* EI    = (const int*)d_in[102];  // [2, 2E]
    const int* srcI  = EI;                     // edge_index[0,:E]
    const int* dstI  = EI + E2;                // edge_index[1,:E]
    const int* inIdx = EI + E2;                // edge_index[1,:]

    // workspace carve (all f32)
    char* w8 = (char*)d_ws;
    size_t off = 0;
    float* e_buf = (float*)(w8 + off); off += (size_t)E2 * 256 * 4;
    float* h_buf = (float*)(w8 + off); off += (size_t)N  * 256 * 4;
    float* agg   = (float*)(w8 + off); off += (size_t)N  * 256 * 4;
    float* inc   = (float*)(w8 + off); off += (size_t)N  * 256 * 4;
    float* hid   = (float*)(w8 + off); off += (size_t)E  * 256 * 4;
    float* z_buf = (float*)(w8 + off); off += (size_t)N  * 512 * 4;
    (void)ws_size; (void)n_in; (void)out_size;

    auto gemm = [&](int M, int K, int K1, int K2,
                    const float* a0, const int* x0, int l0,
                    const float* a1, const int* x1, int l1,
                    const float* a2, const int* x2, int l2,
                    const float* Wp, const float* bp, int flags,
                    const float* g, const float* b,
                    float* outp, const float* res,
                    float* ef, float* eb, const int* dI, const int* sI, float* ag) {
        dim3 grid((M + 63) / 64);
        k_wmma_mlp<<<grid, 256, 0, stream>>>(M, K, K1, K2,
            a0, x0, l0, a1, x1, l1, a2, x2, l2,
            Wp, bp, flags, g, b, outp, res, ef, eb, dI, sI, ag);
    };

    // ---- encoders
    gemm(N, 10, 10, 10, x, 0, 10, 0,0,0, 0,0,0,
         node_enc.W0, node_enc.b0, F_CELU, 0,0, hid, 0, 0,0,0,0,0);
    gemm(N, 256, 256, 256, hid, 0, 256, 0,0,0, 0,0,0,
         node_enc.W1, node_enc.b1, F_LN, node_enc.g, node_enc.b, h_buf, 0, 0,0,0,0,0);
    for (int c = 0; c < 2; ++c) {       // edge encoder chunked (hid sized E)
        gemm(E, 7, 7, 7, edge_attr + (size_t)c * E * 7, 0, 7, 0,0,0, 0,0,0,
             edge_enc.W0, edge_enc.b0, F_CELU, 0,0, hid, 0, 0,0,0,0,0);
        gemm(E, 256, 256, 256, hid, 0, 256, 0,0,0, 0,0,0,
             edge_enc.W1, edge_enc.b1, F_LN, edge_enc.g, edge_enc.b,
             e_buf + (size_t)c * E * 256, 0, 0,0,0,0,0);
    }

    // ---- 6 message-passing layers
    for (int l = 0; l < 6; ++l) {
        hipMemsetAsync(agg, 0, (size_t)N * 256 * 4, stream);
        // edge MLP layer 1: concat(e, h[dst], h[src]) @ W0, CELU
        gemm(E, 768, 256, 512,
             e_buf, 0, 256, h_buf, dstI, 256, h_buf, srcI, 256,
             mp_e[l].W0, mp_e[l].b0, F_CELU, 0,0, hid, 0, 0,0,0,0,0);
        // edge MLP layer 2: LN -> msg; fused e+=msg / e-=msg and atomic agg scatter
        gemm(E, 256, 256, 256, hid, 0, 256, 0,0,0, 0,0,0,
             mp_e[l].W1, mp_e[l].b1, F_LN | F_EDGE | F_SCAT, mp_e[l].g, mp_e[l].b,
             0, 0, e_buf, e_buf + (size_t)E * 256, dstI, srcI, agg);
        // node MLP layer 1: concat(h, agg) @ W0, CELU
        gemm(N, 512, 256, 512, h_buf, 0, 256, agg, 0, 256, 0,0,0,
             mp_n[l].W0, mp_n[l].b0, F_CELU, 0,0, hid, 0, 0,0,0,0,0);
        // node MLP layer 2: LN, residual h = h + out
        gemm(N, 256, 256, 256, hid, 0, 256, 0,0,0, 0,0,0,
             mp_n[l].W1, mp_n[l].b1, F_LN | F_RES, mp_n[l].g, mp_n[l].b,
             h_buf, h_buf, 0,0,0,0,0);
    }

    // ---- readout
    hipMemsetAsync(inc, 0, (size_t)N * 256 * 4, stream);
    k_scatter_rows<<<dim3((E2 + 7) / 8), 256, 0, stream>>>(e_buf, inIdx, inc, E2);
    k_final_ln<<<dim3(N), 256, 0, stream>>>(h_buf, inc, fin_g, fin_b, z_buf, N);

    const float* masks[3] = { bc_disp, bc_disp, bc_rot };
    for (int c = 0; c < 3; ++c) {
        gemm(N, 512, 512, 512, z_buf, 0, 512, 0,0,0, 0,0,0,
             dec[c].W0, dec[c].b0, F_CELU, 0,0, hid, 0, 0,0,0,0,0);
        k_dec_dot<<<dim3((N + 7) / 8), 256, 0, stream>>>(
            hid, dec[c].W1, dec[c].b1, masks[c], (float*)d_out, N, c);
    }
}
